// GAHE_66486093742563
// MI455X (gfx1250) — matrix operand order repaired and checked by reading, not compiled, not run on hardware
//
#include <hip/hip_runtime.h>
#include <hip/hip_bf16.h>

typedef __attribute__((ext_vector_type(16))) __bf16 v16bf;
typedef __attribute__((ext_vector_type(8)))  __bf16 v8bf;
typedef __attribute__((ext_vector_type(8)))  float  v8f;
typedef int v4i __attribute__((__vector_size__(16)));

union Bf16Frag { v16bf v; v8bf h[2]; };
union F4 { float4 v; float a[4]; };

#define BM 128
#define BN 128
#define KT 32
#define LDS_STRIDE 40   // padded row stride (bf16 elems): spreads banks, keeps 16B align

#define SQRT_C 0.316227766016838f
#define CC 0.1f

#define FLAG_RELU 1
#define FLAG_BF16OUT 2

// -------- CDNA5 async global->LDS (guarded; falls back to load+ds_store) --------
#if defined(__gfx1250__) && __has_builtin(__builtin_amdgcn_global_load_async_to_lds_b128)
#define ASYNC_LDS 1
#else
#define ASYNC_LDS 0
#endif

typedef __attribute__((address_space(1))) v4i* gas_v4i_ptr;
typedef __attribute__((address_space(3))) v4i* las_v4i_ptr;

__device__ __forceinline__ void copy16_g2l(const void* g, void* l) {
#if ASYNC_LDS
  __builtin_amdgcn_global_load_async_to_lds_b128(
      (gas_v4i_ptr)(unsigned long long)(uintptr_t)g,
      (las_v4i_ptr)(unsigned)(uintptr_t)l, 0, 0);
#else
  *(uint4*)l = *(const uint4*)g;
#endif
}

__device__ __forceinline__ void wait_async() {
#if ASYNC_LDS
#if __has_builtin(__builtin_amdgcn_s_wait_asynccnt)
  __builtin_amdgcn_s_wait_asynccnt(0);
#else
  asm volatile("s_wait_asynccnt 0x0" ::: "memory");
#endif
#endif
}

// ---------------- scalar helpers ----------------
__device__ __forceinline__ unsigned pack_bf16(float a, float b) {
  union { float f; unsigned u; } x, y;
  x.f = a; y.f = b;
  unsigned xu = x.u + 0x7fffu + ((x.u >> 16) & 1u);  // round-to-nearest-even
  unsigned yu = y.u + 0x7fffu + ((y.u >> 16) & 1u);
  return (xu >> 16) | (yu & 0xffff0000u);
}
__device__ __forceinline__ unsigned short bf16r(float f) {
  union { float f; unsigned u; } v; v.f = f;
  unsigned r = v.u + 0x7fffu + ((v.u >> 16) & 1u);
  return (unsigned short)(r >> 16);
}
__device__ __forceinline__ float bf16tof(unsigned short h) {
  union { unsigned u; float f; } v; v.u = ((unsigned)h) << 16; return v.f;
}
__device__ __forceinline__ float sigmoidf(float v) { return 1.f / (1.f + __expf(-v)); }

// blockDim.x == 128 (4 waves). All threads return the block-wide sum.
__device__ __forceinline__ float rowsum(float v, float* sm) {
#pragma unroll
  for (int o = 16; o > 0; o >>= 1) v += __shfl_xor(v, o, 32);
  __syncthreads();
  if ((threadIdx.x & 31) == 0) sm[threadIdx.x >> 5] = v;
  __syncthreads();
  return sm[0] + sm[1] + sm[2] + sm[3];
}

__device__ __forceinline__ float fragDot(const float* a, const float* b, float* sm) {
  float s = 0.f;
#pragma unroll
  for (int i = 0; i < 4; ++i) s += a[i] * b[i];
  return rowsum(s, sm);
}

template <bool SPH>
__device__ __forceinline__ void expmapN(const float* u, float* o, float* sm) {
  float n = fmaxf(sqrtf(fragDot(u, u, sm)), 1e-15f);
  float a = SQRT_C * n;
  float f = (SPH ? tanf(a) : tanhf(a)) / a;
#pragma unroll
  for (int i = 0; i < 4; ++i) o[i] = f * u[i];
}

template <bool SPH>
__device__ __forceinline__ void logmapN(const float* y, float* o, float* sm) {
  float n = fmaxf(sqrtf(fragDot(y, y, sm)), 1e-15f);
  float a = SQRT_C * n;
  float f;
  if (SPH) {
    f = atanf(a) / a;
  } else {
    float t = fminf(fmaxf(a, -1.f + 1e-7f), 1.f - 1e-7f);
    f = 0.5f * __logf((1.f + t) / (1.f - t)) / a;   // artanh
  }
#pragma unroll
  for (int i = 0; i < 4; ++i) o[i] = f * y[i];
}

__device__ __forceinline__ void mobius(const float* xv, const float* yv, float ysign,
                                       float* o, float* sm) {
  float x2 = fragDot(xv, xv, sm);
  float y2 = fragDot(yv, yv, sm);
  float xy = ysign * fragDot(xv, yv, sm);
  float ca = 1.f + 2.f * CC * xy + CC * y2;
  float cb = 1.f - CC * x2;
  float den = 1.f + 2.f * CC * xy + CC * CC * x2 * y2;
  float inv = 1.f / fmaxf(den, 1e-15f);
#pragma unroll
  for (int i = 0; i < 4; ++i) o[i] = (ca * xv[i] + cb * ysign * yv[i]) * inv;
}

template <bool SPH>
__device__ void geo(const float* p, const float* q, float* out, float* sm) {
  float hx[4], qx[4], m[4], d[4], e[4], q1[4];
  expmapN<SPH>(p, hx, sm);
  expmapN<SPH>(q, qx, sm);
  mobius(hx, qx, -1.f, m, sm);
  logmapN<SPH>(m, d, sm);
#pragma unroll
  for (int i = 0; i < 4; ++i) d[i] *= sigmoidf(d[i]);
  expmapN<SPH>(d, e, sm);
  mobius(qx, e, 1.f, q1, sm);
  mobius(q1, hx, -1.f, m, sm);
  logmapN<SPH>(m, d, sm);
#pragma unroll
  for (int i = 0; i < 4; ++i) d[i] *= sigmoidf(d[i]);
  expmapN<SPH>(d, e, sm);
  mobius(hx, e, 1.f, m, sm);
  logmapN<SPH>(m, out, sm);
}

// ---------------- kernels ----------------

// Gather P (f32 + bf16 copies), fac_rel, fac_rhs.  grid=1000, block=128
__global__ void prep_kernel(const int* __restrict__ x, const float* __restrict__ ent,
                            const float* __restrict__ rel, float* __restrict__ Pf,
                            unsigned short* __restrict__ Pb,
                            float* __restrict__ facRel, float* __restrict__ facRhs) {
  int b = blockIdx.x;
  int t = threadIdx.x;
  int h  = x[b * 3 + 0];
  int r  = x[b * 3 + 1];
  int tl = x[b * 3 + 2];
  const float4* eh = (const float4*)(ent + (size_t)h * 1024);
  const float4* er = (const float4*)(rel + (size_t)r * 1024);
  const float4* et = (const float4*)(ent + (size_t)tl * 1024);
  float4 hr = eh[t], hi = eh[t + 128];
  ((float4*)(Pf + (size_t)b * 512))[t] = hr;
  ((float4*)(Pf + (size_t)(1000 + b) * 512))[t] = hi;
  ((uint2*)(Pb + (size_t)b * 512))[t] =
      make_uint2(pack_bf16(hr.x, hr.y), pack_bf16(hr.z, hr.w));
  ((uint2*)(Pb + (size_t)(1000 + b) * 512))[t] =
      make_uint2(pack_bf16(hi.x, hi.y), pack_bf16(hi.z, hi.w));
  F4 rr, ri, tr, ti, fo;
  rr.v = er[t]; ri.v = er[t + 128];
#pragma unroll
  for (int i = 0; i < 4; ++i) fo.a[i] = sqrtf(rr.a[i] * rr.a[i] + ri.a[i] * ri.a[i]);
  ((float4*)(facRel + (size_t)b * 512))[t] = fo.v;
  tr.v = et[t]; ti.v = et[t + 128];
#pragma unroll
  for (int i = 0; i < 4; ++i) fo.a[i] = sqrtf(tr.a[i] * tr.a[i] + ti.a[i] * ti.a[i]);
  ((float4*)(facRhs + (size_t)b * 512))[t] = fo.v;
}

// Per-row geometry: h_e | h_h | h_s -> CATb[2000,1536] bf16.  grid=2000, block=128
__global__ void branch_kernel(const float* __restrict__ P, const float* __restrict__ Q,
                              unsigned short* __restrict__ CATb) {
  __shared__ float sm[4];
  int row = blockIdx.x;
  int t = threadIdx.x;
  F4 pv, qv;
  pv.v = ((const float4*)(P + (size_t)row * 512))[t];
  qv.v = ((const float4*)(Q + (size_t)row * 512))[t];
  F4 he, hh, hs;
#pragma unroll
  for (int i = 0; i < 4; ++i) {
    float d1 = pv.a[i] - qv.a[i];
    float h1 = d1 * sigmoidf(d1);
    float d2 = qv.a[i] + h1 - pv.a[i];
    he.a[i] = pv.a[i] + d2 * sigmoidf(d2);
  }
  geo<false>(pv.a, qv.a, hh.a, sm);
  geo<true>(pv.a, qv.a, hs.a, sm);
  uint2* cat = (uint2*)(CATb + (size_t)row * 1536);
  cat[t]       = make_uint2(pack_bf16(he.a[0], he.a[1]), pack_bf16(he.a[2], he.a[3]));
  cat[t + 128] = make_uint2(pack_bf16(hh.a[0], hh.a[1]), pack_bf16(hh.a[2], hh.a[3]));
  cat[t + 256] = make_uint2(pack_bf16(hs.a[0], hs.a[1]), pack_bf16(hs.a[2], hs.a[3]));
}

// Gb[:,0:512]=h_I ; Gb[:,512+j]=cat[j]*sigmoid(g).  grid=2000, block=256
__global__ void gate_kernel(const unsigned short* __restrict__ CATb,
                            const float* __restrict__ G, unsigned short* __restrict__ Gb) {
  int row = blockIdx.x;
  for (int j = threadIdx.x; j < 2048; j += 256) {
    float g = G[(size_t)row * 2048 + j];
    float v = (j < 512) ? g : bf16tof(CATb[(size_t)row * 1536 + (j - 512)]) * sigmoidf(g);
    Gb[(size_t)row * 2048 + j] = bf16r(v);
  }
}

// l2norm + residual + complex combine + fac_lhs. A stored bf16.  grid=1000, block=128
__global__ void finalize_kernel(const float* __restrict__ P, const float* __restrict__ F,
                                const int* __restrict__ x, const float* __restrict__ rel,
                                unsigned short* __restrict__ Ab, float* __restrict__ facLhs) {
  __shared__ float sm[4];
  int b = blockIdx.x, t = threadIdx.x;
  F4 fr, fi;
  fr.v = ((const float4*)(F + (size_t)b * 512))[t];
  fi.v = ((const float4*)(F + (size_t)(1000 + b) * 512))[t];
  float s1 = 0.f, s2 = 0.f;
#pragma unroll
  for (int i = 0; i < 4; ++i) { s1 += fr.a[i] * fr.a[i]; s2 += fi.a[i] * fi.a[i]; }
  float n1 = fmaxf(sqrtf(rowsum(s1, sm)), 1e-12f);
  float n2 = fmaxf(sqrtf(rowsum(s2, sm)), 1e-12f);
  F4 pr, pi, rr, ri, ar, ai, fl;
  pr.v = ((const float4*)(P + (size_t)b * 512))[t];
  pi.v = ((const float4*)(P + (size_t)(1000 + b) * 512))[t];
  int r = x[b * 3 + 1];
  rr.v = ((const float4*)(rel + (size_t)r * 1024))[t];
  ri.v = ((const float4*)(rel + (size_t)r * 1024))[t + 128];
#pragma unroll
  for (int i = 0; i < 4; ++i) {
    float lr = 0.5f * pr.a[i] + 0.5f * fr.a[i] / n1;
    float li = 0.5f * pi.a[i] + 0.5f * fi.a[i] / n2;
    ar.a[i] = lr * rr.a[i] - li * ri.a[i];
    ai.a[i] = lr * ri.a[i] + li * rr.a[i];
    fl.a[i] = sqrtf(lr * lr + li * li);
  }
  ((uint2*)(Ab + (size_t)b * 1024))[t] =
      make_uint2(pack_bf16(ar.a[0], ar.a[1]), pack_bf16(ar.a[2], ar.a[3]));
  ((uint2*)(Ab + (size_t)b * 1024 + 512))[t] =
      make_uint2(pack_bf16(ai.a[0], ai.a[1]), pack_bf16(ai.a[2], ai.a[3]));
  ((float4*)(facLhs + (size_t)b * 512))[t] = fl.v;
}

// C[M,N] = A[M,K](bf16) @ B[N,K](f32->bf16)^T (+bias, relu / bf16-out flags).
// 256 threads = 8 waves (4 M-strips x 2 N-strips); wave tile 32x64 = 8 WMMAs/chunk.
// A tiles: async global->LDS (bf16, no conversion). B tiles: f32 -> regs -> bf16 -> LDS.
__global__ __launch_bounds__(256) void gemm_bf16_kernel(
    const unsigned short* __restrict__ A, int lda, const float* __restrict__ B, int ldb,
    const float* __restrict__ bias, void* __restrict__ Cv, int ldc,
    int M, int N, int K, int flags) {
  __shared__ __bf16 As[2][BM * LDS_STRIDE];
  __shared__ __bf16 Bs[2][BN * LDS_STRIDE];

  const int tid  = threadIdx.x;
  const int lane = tid & 31;
  const int wave = tid >> 5;
  const int wm = wave & 3;   // M strip of 32
  const int wn = wave >> 2;  // N strip of 64
  const int mBase = blockIdx.x * BM;
  const int nBase = blockIdx.y * BN;

  // A async staging: each thread copies two 16B chunks (rows r0, r0+64)
  const int ar0 = tid >> 2;
  const int ar1 = ar0 + 64;
  const int ac  = (tid & 3) * 8;  // bf16 elems
  int am0 = mBase + ar0; if (am0 > M - 1) am0 = M - 1;
  int am1 = mBase + ar1; if (am1 > M - 1) am1 = M - 1;
  const size_t aoff0 = (size_t)am0 * lda + ac;
  const size_t aoff1 = (size_t)am1 * lda + ac;

  // B staging: each thread loads 16 consecutive f32 of one row
  const int br = tid >> 1;
  const int bc = (tid & 1) * 16;
  int bn = nBase + br; if (bn > N - 1) bn = N - 1;
  const size_t boff = (size_t)bn * ldb + bc;

  unsigned bpk[8];
  auto issueA = [&](int buf, int k0) {
    copy16_g2l(A + aoff0 + k0, &As[buf][ar0 * LDS_STRIDE + ac]);
    copy16_g2l(A + aoff1 + k0, &As[buf][ar1 * LDS_STRIDE + ac]);
  };
  auto loadB = [&](int k0) {
    const float4* p = (const float4*)(B + boff + k0);
#pragma unroll
    for (int i = 0; i < 4; ++i) {
      float4 v = p[i];
      bpk[2 * i]     = pack_bf16(v.x, v.y);
      bpk[2 * i + 1] = pack_bf16(v.z, v.w);
    }
  };
  auto storeB = [&](int buf) {
    uint4* d = (uint4*)&Bs[buf][br * LDS_STRIDE + bc];
    d[0] = make_uint4(bpk[0], bpk[1], bpk[2], bpk[3]);
    d[1] = make_uint4(bpk[4], bpk[5], bpk[6], bpk[7]);
  };

  v8f acc[2][4] = {};
  const int nk = K / KT;

  issueA(0, 0);
  loadB(0);
  storeB(0);
  wait_async();
  int cur = 0;
  for (int kt = 0; kt < nk; ++kt) {
    __syncthreads();
    const bool more = (kt + 1 < nk);
    if (more) { issueA(cur ^ 1, (kt + 1) * KT); loadB((kt + 1) * KT); }
    if (kt + 2 < nk) __builtin_prefetch(B + boff + (size_t)(kt + 2) * KT, 0, 1);

    const int ln = lane & 15;
    const int kb16 = (lane >> 4) << 4;
    const int kb8  = (lane >> 4) << 3;  // A: lanes 16-31 hold K={8..15,24..31}
    Bf16Frag bf[4];
#pragma unroll
    for (int in = 0; in < 4; ++in) {
      const __bf16* bb = &Bs[cur][(wn * 64 + in * 16 + ln) * LDS_STRIDE + kb8];
      bf[in].h[0] = *(const v8bf*)bb;
      bf[in].h[1] = *(const v8bf*)(bb + 16);
    }
#pragma unroll
    for (int im = 0; im < 2; ++im) {
      Bf16Frag af;
      const __bf16* ab = &As[cur][(wm * 32 + im * 16 + ln) * LDS_STRIDE + kb8];
      af.h[0] = *(const v8bf*)ab;
      af.h[1] = *(const v8bf*)(ab + 16);
#pragma unroll
      for (int in = 0; in < 4; ++in)
        acc[im][in] = __builtin_amdgcn_wmma_f32_16x16x32_bf16(
            false, af.v, false, bf[in].v, (short)0, acc[im][in], false, false);
    }
    if (more) { storeB(cur ^ 1); wait_async(); }
    cur ^= 1;
  }

  // epilogue: VGPR r -> M=r (lanes 0-15) / M=r+8 (lanes 16-31); N = lane%16
  const int cm = (lane >> 4) * 8;
  const int cn = lane & 15;
#pragma unroll
  for (int im = 0; im < 2; ++im)
#pragma unroll
    for (int in = 0; in < 4; ++in) {
      int col = nBase + wn * 64 + in * 16 + cn;
      if (col >= N) continue;
      float bi = bias ? bias[col] : 0.f;
#pragma unroll
      for (int r = 0; r < 8; ++r) {
        int row = mBase + wm * 32 + im * 16 + cm + r;
        if (row < M) {
          float v = acc[im][in][r] + bi;
          if (flags & FLAG_RELU) v = fmaxf(v, 0.f);
          if (flags & FLAG_BF16OUT)
            ((unsigned short*)Cv)[(size_t)row * ldc + col] = bf16r(v);
          else
            ((float*)Cv)[(size_t)row * ldc + col] = v;
        }
      }
    }
}

// ---------------- host ----------------
extern "C" void kernel_launch(void* const* d_in, const int* in_sizes, int n_in,
                              void* d_out, int out_size, void* d_ws, size_t ws_size,
                              hipStream_t stream) {
  (void)in_sizes; (void)n_in; (void)out_size; (void)ws_size;
  const int*   x    = (const int*)d_in[0];
  const float* ent  = (const float*)d_in[1];
  const float* rel  = (const float*)d_in[2];
  const float* w1   = (const float*)d_in[3];
  const float* b1   = (const float*)d_in[4];
  const float* w2   = (const float*)d_in[5];
  const float* b2   = (const float*)d_in[6];
  const float* wi   = (const float*)d_in[7];
  const float* wib  = (const float*)d_in[8];
  const float* we   = (const float*)d_in[9];
  const float* web  = (const float*)d_in[10];
  const float* wh   = (const float*)d_in[11];
  const float* whb  = (const float*)d_in[12];
  const float* wsp  = (const float*)d_in[13];
  const float* wspb = (const float*)d_in[14];
  const float* wu   = (const float*)d_in[15];
  const float* wub  = (const float*)d_in[16];

  float* out = (float*)d_out;
  char*  wsb = (char*)d_ws;
  float* Pf = (float*)(wsb + 0);          // [2000,512] f32
  float* Q  = (float*)(wsb + 4096000);    // [2000,512] f32
  float* F  = (float*)(wsb + 8192000);    // [2000,512] f32 (fused)
  float* G  = (float*)(wsb + 12288000);   // [2000,2048] f32
  unsigned short* Pb   = (unsigned short*)(wsb + 28672000);  // [2000,512]
  unsigned short* Hb   = (unsigned short*)(wsb + 30720000);  // [2000,512]
  unsigned short* CATb = (unsigned short*)(wsb + 32768000);  // [2000,1536]
  unsigned short* Gb   = (unsigned short*)(wsb + 38912000);  // [2000,2048]
  unsigned short* Ab   = (unsigned short*)(wsb + 47104000);  // [1000,1024]

  float* score  = out;
  float* facLhs = out + 100000000ll;
  float* facRel = out + 100512000ll;
  float* facRhs = out + 101024000ll;

  auto gemm = [&](const unsigned short* Ap, int lda, const float* Bp, int ldb,
                  const float* bp, void* Cp, int ldc, int M, int N, int K, int flags) {
    dim3 grid((M + BM - 1) / BM, (N + BN - 1) / BN);
    gemm_bf16_kernel<<<grid, 256, 0, stream>>>(Ap, lda, Bp, ldb, bp, Cp, ldc, M, N, K, flags);
  };

  prep_kernel<<<1000, 128, 0, stream>>>(x, ent, rel, Pf, Pb, facRel, facRhs);
  gemm(Pb, 512, w1, 512, b1, Hb, 512, 2000, 512, 512, FLAG_RELU | FLAG_BF16OUT);
  gemm(Hb, 512, w2, 512, b2, Q, 512, 2000, 512, 512, 0);
  branch_kernel<<<2000, 128, 0, stream>>>(Pf, Q, CATb);
  gemm(CATb, 1536, wi,  1536, wib,  G,        2048, 2000, 512, 1536, 0);
  gemm(CATb, 1536, we,  1536, web,  G + 512,  2048, 2000, 512, 1536, 0);
  gemm(CATb, 1536, wh,  1536, whb,  G + 1024, 2048, 2000, 512, 1536, 0);
  gemm(CATb, 1536, wsp, 1536, wspb, G + 1536, 2048, 2000, 512, 1536, 0);
  gate_kernel<<<2000, 256, 0, stream>>>(CATb, G, Gb);
  gemm(Gb, 2048, wu, 2048, wub, F, 512, 2000, 512, 2048, 0);
  finalize_kernel<<<1000, 128, 0, stream>>>(Pf, F, x, rel, Ab, facLhs);
  gemm(Ab, 1024, ent, 1024, nullptr, score, 100000, 1000, 100000, 1024, 0);
}
